// W2NER_66537633350296
// MI455X (gfx1250) — matrix operand order repaired and checked by reading, not compiled, not run on hardware
//
#include <hip/hip_runtime.h>
#include <hip/hip_bf16.h>

typedef __attribute__((ext_vector_type(16))) _Float16 v16h;
typedef __attribute__((ext_vector_type(8)))  float    v8f;

#define Bsz 4
#define Nn  512
#define Dd  128
#define Cc  10
#define JT_PER_WAVE 8       // 8 tiles of 16 rows -> 128 j per wave
#define WAVES_PER_BLOCK 8

// ws accumulators: [0]=tp [1]=tn [2]=fp [3]=loss_sum [4]=acc_sum
__global__ void w2ner_zero(float* acc) {
    if (threadIdx.x < 8) acc[threadIdx.x] = 0.0f;
}

__global__ __launch_bounds__(256)
void w2ner_main(const float* __restrict__ x, const int* __restrict__ span,
                const int* __restrict__ seqlen, const float* __restrict__ W,
                const float* __restrict__ bias, float* __restrict__ out,
                float* __restrict__ acc)
{
    __shared__ float lds[WAVES_PER_BLOCK][16][16];

    const int lane  = threadIdx.x & 31;
    const int wv    = threadIdx.x >> 5;
    const int gwave = blockIdx.x * WAVES_PER_BLOCK + wv;

    const int chunks = (Nn / 16) / JT_PER_WAVE;   // 4 chunks of 128 j
    const int jchunk = gwave % chunks;
    const int bi     = gwave / chunks;
    const int i      = bi % Nn;
    const int b      = bi / Nn;

    const float* xb = x + (size_t)b * Nn * Dd;
    const float* xi = xb + (size_t)i * Dd;

    const int col = lane & 15;            // B column (class c) / A row-in-tile / D column
    const int hi  = (lane >> 4);          // 0 for lanes 0-15, 1 for lanes 16-31
    const int hiK = hi << 3;              // A-layout K offset (0 or 8)

    // ---- Build B fragments once per wave: Bf[d][c] = x_i[d] * W[d][c] (f16) ----
    // B layout (16-bit, 32x16): lane holds column n = lane&15; half h -> K = (h&7) + 8*hi + 16*(h>>3)
    v16h bf[4];
    for (int ks = 0; ks < 4; ++ks) {
        const int k0 = ks * 32;
        for (int h = 0; h < 16; ++h) {
            const int K = (h & 7) + (hi << 3) + ((h >> 3) << 4);
            float v = 0.0f;
            if (col < Cc) v = xi[k0 + K] * W[(k0 + K) * Cc + col];
            bf[ks][h] = (_Float16)v;
        }
    }

    const int sl    = seqlen[b];
    const bool irow = (i < sl);

    float p_tp = 0.f, p_tn = 0.f, p_fp = 0.f, p_loss = 0.f, p_acc = 0.f;

    for (int jt = 0; jt < JT_PER_WAVE; ++jt) {
        const int jbase = (jchunk * JT_PER_WAVE + jt) * 16;

        // ---- A fragment: 16 rows of x_j (f16). Row M = lane&15.
        // A layout (16-bit, 16x32): half h -> K = (h&7) + 8*hi + 16*(h>>3)
        const float* xj = xb + (size_t)(jbase + col) * Dd;

        v8f accv = {};
        for (int ks = 0; ks < 4; ++ks) {
            const int k0 = ks * 32;
            v16h a;
            for (int h = 0; h < 16; ++h) {
                const int K = (h & 7) + hiK + ((h >> 3) << 4);
                a[h] = (_Float16)xj[k0 + K];
            }
            accv = __builtin_amdgcn_wmma_f32_16x16x32_f16(
                false, a, false, bf[ks], (short)0, accv, false, false);
        }

        // ---- Lane-transpose the 16x16 tile through LDS ----
        // D layout: element (m,n): vgpr v, lane L -> m = v + 8*(L>=16), n = L&15
        {
            const int mbase = hi << 3;
            #pragma unroll
            for (int v = 0; v < 8; ++v)
                lds[wv][mbase + v][col] = accv[v];
        }
        asm volatile("s_wait_dscnt 0" ::: "memory");   // cross-lane DS dependency

        // ---- Epilogue: lanes 0..15 each own one row j ----
        float l_loss = 0.f, l_acc = 0.f, l_tp = 0.f, l_tn = 0.f, l_fp = 0.f;
        if (lane < 16) {
            const int j = jbase + lane;
            const int t = span[((size_t)b * Nn + i) * Nn + j];

            float lg[Cc];
            float mx = -1e30f;
            #pragma unroll
            for (int c = 0; c < Cc; ++c) {
                lg[c] = lds[wv][lane][c] + bias[c];
                mx = fmaxf(mx, lg[c]);
            }
            float se = 0.f;
            int   am = 0;
            float bm = lg[0];
            #pragma unroll
            for (int c = 0; c < Cc; ++c) {
                se += __expf(lg[c] - mx);
                if (lg[c] > bm) { bm = lg[c]; am = c; }
            }
            const float p = __expf(lg[t] - mx) / se;

            const bool val = irow && (j < sl);
            const float vf = val ? 1.f : 0.f;
            const int pred = val ? am : 0;

            out[((size_t)b * Nn + i) * Nn + j] = (float)pred;

            const float om = 1.f - p;
            l_loss = -(om * om) * __logf(p) * vf;                    // focal, GAMMA=2
            l_acc  = (pred == t) ? vf : 0.f;
            l_tp   = ((pred == t) && (t > 0)) ? 1.f : 0.f;
            l_tn   = ((pred != t) && (t > 0)) ? 1.f : 0.f;
            l_fp   = ((pred > 0) && (t <= 0) && val) ? 1.f : 0.f;
        }
        asm volatile("" ::: "memory");   // keep next tile's LDS stores after reads

        p_loss += l_loss; p_acc += l_acc; p_tp += l_tp; p_tn += l_tn; p_fp += l_fp;
    }

    // ---- wave32 reduction, one atomic set per wave ----
    #pragma unroll
    for (int off = 16; off > 0; off >>= 1) {
        p_tp   += __shfl_down(p_tp,   off);
        p_tn   += __shfl_down(p_tn,   off);
        p_fp   += __shfl_down(p_fp,   off);
        p_loss += __shfl_down(p_loss, off);
        p_acc  += __shfl_down(p_acc,  off);
    }
    if (lane == 0) {
        atomicAdd(&acc[0], p_tp);
        atomicAdd(&acc[1], p_tn);
        atomicAdd(&acc[2], p_fp);
        atomicAdd(&acc[3], p_loss);
        atomicAdd(&acc[4], p_acc);
    }
}

__global__ void w2ner_fin(const float* __restrict__ acc,
                          const int* __restrict__ seqlen,
                          float* __restrict__ out_tail)
{
    if (threadIdx.x == 0) {
        float s2 = 0.f;
        for (int b = 0; b < Bsz; ++b) {
            const float s = (float)seqlen[b];
            s2 += s * s;
        }
        out_tail[0] = acc[0];            // tp
        out_tail[1] = acc[1];            // tn
        out_tail[2] = acc[2];            // fp
        out_tail[3] = acc[3] / s2;       // loss
        out_tail[4] = acc[4] / s2;       // accuracy
    }
}

extern "C" void kernel_launch(void* const* d_in, const int* in_sizes, int n_in,
                              void* d_out, int out_size, void* d_ws, size_t ws_size,
                              hipStream_t stream) {
    const float* x      = (const float*)d_in[0];   // [4,512,128]
    const int*   span   = (const int*)  d_in[1];   // [4,512,512]
    const int*   seqlen = (const int*)  d_in[2];   // [4]
    const float* W      = (const float*)d_in[3];   // [128,10]
    const float* bias   = (const float*)d_in[4];   // [10]
    float* out = (float*)d_out;                    // predict map + 5 scalars
    float* acc = (float*)d_ws;

    (void)in_sizes; (void)n_in; (void)out_size; (void)ws_size;

    w2ner_zero<<<1, 64, 0, stream>>>(acc);

    const int chunks      = (Nn / 16) / JT_PER_WAVE;       // 4
    const int total_waves = Bsz * Nn * chunks;             // 8192
    const int blocks      = total_waves / WAVES_PER_BLOCK; // 1024
    w2ner_main<<<blocks, 256, 0, stream>>>(x, span, seqlen, W, bias, out, acc);

    w2ner_fin<<<1, 32, 0, stream>>>(acc, seqlen, out + (size_t)Bsz * Nn * Nn);
}
